// AttentionNaive_40793599377930
// MI455X (gfx1250) — matrix operand order repaired and checked by reading, not compiled, not run on hardware
//
#include <hip/hip_runtime.h>

// ---------------------------------------------------------------------------
// MHA forward for MI455X (gfx1250, wave32, WMMA).
// S=4096, D=2048, H=16, HD=128.  Compute-bound: everything runs through
// v_wmma_f32_16x16x32_bf16 with fp32 accumulation.
//  - GEMMs: double-buffered LDS tiles via GLOBAL_LOAD_ASYNC_TO_LDS_B128
//  - Attention: K tiles via the Tensor Data Mover (TENSOR_LOAD_TO_LDS + D#)
// ---------------------------------------------------------------------------

#define SEQ     4096
#define DMODEL  2048
#define NHEAD   16
#define HDIM    128

typedef __attribute__((ext_vector_type(16))) __bf16        v16bf;
typedef __attribute__((ext_vector_type(8)))  float         v8f;
typedef __attribute__((ext_vector_type(4)))  unsigned int  v4u;
typedef __attribute__((ext_vector_type(8)))  unsigned int  v8u;
typedef __attribute__((ext_vector_type(4)))  float         v4f;

union FragB16 {            // one WMMA bf16 A/B operand: 16 bf16 per lane = 8 VGPRs
    v16bf v;
    v4u   u[2];
};

#define LDS_OFF(p) ((unsigned int)(unsigned long long) \
    (__attribute__((address_space(3))) unsigned short*)(p))

// Async global->LDS b128 copy (GLOBAL_LOAD_ASYNC_TO_LDS_B128, GV mode).
__device__ __forceinline__ void copy_b128_to_lds(const unsigned short* g,
                                                 unsigned short* l) {
    unsigned int lds_off = LDS_OFF(l);
    asm volatile("global_load_async_to_lds_b128 %0, %1, off"
                 :
                 : "v"(lds_off), "v"(g)
                 : "memory");
}

__device__ __forceinline__ void wait_async() {
    asm volatile("s_wait_asynccnt 0x0" ::: "memory");
}

// Tensor Data Mover: 2D tile (tile_d0 x tile_d1, 8-byte elements) from a
// row-major tensor into LDS.  Descriptor per CDNA5 ISA D# groups 0/1.
// g1w0_flags packs data_size / pad_enable / pad_interval / pad_amount.
__device__ __forceinline__ void tdm_load_2d_to_lds(
    const void* gaddr, unsigned int lds_off,
    unsigned int tensor_d0, unsigned int tensor_d1, unsigned int d0_stride,
    unsigned int tile_d0, unsigned int tile_d1, unsigned int g1w0_flags)
{
    unsigned long long ga = (unsigned long long)gaddr;
    v4u g0;
    g0.x = 1u;                                                    // count=1 (user)
    g0.y = lds_off;                                               // LDS byte addr
    g0.z = (unsigned int)ga;                                      // global addr lo
    g0.w = (unsigned int)((ga >> 32) & 0x01ffffffu) | (2u << 30); // addr hi | type=2
    v8u g1;
    g1.s0 = g1w0_flags;                                   // wg_mask=0 | data_size | pad
    g1.s1 = (tensor_d0 & 0xffffu) << 16;                  // barrier=0 | dim0[15:0]
    g1.s2 = (tensor_d0 >> 16) | ((tensor_d1 & 0xffffu) << 16);
    g1.s3 = (tensor_d1 >> 16) | (tile_d0 << 16);          // dim1 hi | tile_dim0
    g1.s4 = tile_d1;                                      // tile_dim1 | tile_dim2=0
    g1.s5 = d0_stride;                                    // dim0_stride[31:0]
    g1.s6 = 0u;                                           // stride hi | dim1_stride lo
    g1.s7 = 0u;
    asm volatile("tensor_load_to_lds %0, %1"
                 :
                 : "s"(g0), "s"(g1)
                 : "memory");
}

__device__ __forceinline__ unsigned short f2bf(float f) {
    unsigned int u = __float_as_uint(f);
    u += 0x7fffu + ((u >> 16) & 1u);        // round to nearest even
    return (unsigned short)(u >> 16);
}

__device__ __forceinline__ v8f wmma_bf16(const FragB16& a, const FragB16& b, v8f c) {
    return __builtin_amdgcn_wmma_f32_16x16x32_bf16(
        /*neg_a=*/false, a.v, /*neg_b=*/false, b.v,
        /*c_mod=*/(short)0, c, /*reuse_a=*/false, /*reuse_b=*/false);
}

// ---------------------------------------------------------------------------
// fp32 -> bf16 conversion (bandwidth-trivial; feeds the WMMA pipelines)
// ---------------------------------------------------------------------------
__global__ __launch_bounds__(256) void f32_to_bf16_kernel(
    const float* __restrict__ in, unsigned short* __restrict__ out, int n)
{
    int i = (blockIdx.x * 256 + threadIdx.x) * 4;
    if (i < n) {
        v4f f = *(const v4f*)(in + i);
        unsigned int lo = (unsigned int)f2bf(f.x) | ((unsigned int)f2bf(f.y) << 16);
        unsigned int hi = (unsigned int)f2bf(f.z) | ((unsigned int)f2bf(f.w) << 16);
        *(unsigned int*)(out + i)     = lo;
        *(unsigned int*)(out + i + 2) = hi;
    }
}

// ---------------------------------------------------------------------------
// C[M,N] = A[M,K] * B[N,K]^T + bias[N].
// A, B bf16 row-major. OUTF=0 -> bf16 out, OUTF=1 -> fp32 out.
// Block: 256 threads (8 waves). Tile: 128 rows x 128 cols.
// Double-buffered 128x32 B tile in LDS via async-to-LDS loads.
// ---------------------------------------------------------------------------
template <int OUTF>
__global__ __launch_bounds__(256) void gemm_bt_bias(
    const unsigned short* __restrict__ A,
    const unsigned short* __restrict__ B,
    const float* __restrict__ bias,
    void* __restrict__ Cout, int M, int N, int K)
{
    __shared__ unsigned short Bs[2][128][48];   // 32 K-elems + pad; 96B row stride

    const int lane = threadIdx.x & 31;
    const int wave = threadIdx.x >> 5;
    const int r0   = blockIdx.y * 128 + wave * 16;
    const int n0   = blockIdx.x * 128;

    v8f acc[8];
#pragma unroll
    for (int s = 0; s < 8; ++s)
#pragma unroll
        for (int i = 0; i < 8; ++i) acc[s][i] = 0.0f;

    // issue async copy of one 128x32 B slice into buffer `buf`
    auto issue_tile = [&](int k0, int buf) {
#pragma unroll
        for (int q = 0; q < 2; ++q) {
            int c   = threadIdx.x + q * 256;   // 512 b128 chunks: 128 rows x 4
            int row = c >> 2;
            int kp  = (c & 3) << 3;
            copy_b128_to_lds(B + (size_t)(n0 + row) * K + k0 + kp, &Bs[buf][row][kp]);
        }
    };

    // A-matrix fragment base: lane&15 = M row; lane>>4 selects K 0-7/16-23 vs 8-15/24-31
    const unsigned short* Aptr = A + (size_t)(r0 + (lane & 15)) * K + ((lane >> 4) << 3);

    issue_tile(0, 0);
    wait_async();
    __syncthreads();

    for (int k0 = 0; k0 < K; k0 += 32) {
        const int buf = (k0 >> 5) & 1;
        if (k0 + 32 < K) issue_tile(k0 + 32, buf ^ 1);   // prefetch next slice

        FragB16 af;
        af.u[0] = *(const v4u*)(Aptr + k0);
        af.u[1] = *(const v4u*)(Aptr + k0 + 16);

#pragma unroll
        for (int s = 0; s < 8; ++s) {
            FragB16 bf_;
            const unsigned short* bp = &Bs[buf][s * 16 + (lane & 15)][(lane >> 4) << 3];
            bf_.u[0] = *(const v4u*)(bp);
            bf_.u[1] = *(const v4u*)(bp + 16);
            acc[s] = wmma_bf16(af, bf_, acc[s]);
        }

        wait_async();        // next slice landed in LDS
        __syncthreads();     // all waves done reading current slice
    }

    const int rb = r0 + ((lane >> 4) << 3);
#pragma unroll
    for (int s = 0; s < 8; ++s) {
        int   col = n0 + s * 16 + (lane & 15);
        float bv  = bias[col];
#pragma unroll
        for (int i = 0; i < 8; ++i) {
            float v = acc[s][i] + bv;
            if (OUTF) ((float*)Cout)[(size_t)(rb + i) * N + col] = v;
            else      ((unsigned short*)Cout)[(size_t)(rb + i) * N + col] = f2bf(v);
        }
    }
}

// ---------------------------------------------------------------------------
// Causal flash attention. Grid: (SEQ/64, NHEAD), block = 128 (4 waves).
// Wave w owns 16 query rows; workgroup streams 64-key blocks through LDS.
// K tile is fetched by the TDM (one tensor_load_to_lds from wave 0, with the
// descriptor's pad fields reproducing the +16B row padding of Ks); the DMA
// overlaps the all-thread V-transpose scatter.  TENSORcnt gates consumption.
// ---------------------------------------------------------------------------
__global__ __launch_bounds__(128) void attn_fwd(
    const unsigned short* __restrict__ Q,
    const unsigned short* __restrict__ K,
    const unsigned short* __restrict__ V,
    unsigned short* __restrict__ O)
{
    __shared__ unsigned short Ks[64][HDIM + 8];   // K tile, row-major [key][dim]
    __shared__ unsigned short Vt[HDIM][64 + 8];   // V tile transposed [dim][key]
    __shared__ unsigned short Ps[4][16][72];      // per-wave P tile [qrow][key]

    const int lane = threadIdx.x & 31;
    const int wave = threadIdx.x >> 5;
    const int h    = blockIdx.y;
    const int qb   = blockIdx.x;
    const int qr0  = qb * 64 + wave * 16;
    const int hoff = h * HDIM;

    // Preload Q fragments (16 rows x 128 dims = 4 A-fragments of K=32)
    FragB16 qf[4];
    {
        const unsigned short* qp =
            Q + (size_t)(qr0 + (lane & 15)) * DMODEL + hoff + ((lane >> 4) << 3);
#pragma unroll
        for (int t = 0; t < 4; ++t) {
            qf[t].u[0] = *(const v4u*)(qp + t * 32);
            qf[t].u[1] = *(const v4u*)(qp + t * 32 + 16);
        }
    }

    v8f o[8];
#pragma unroll
    for (int n = 0; n < 8; ++n)
#pragma unroll
        for (int i = 0; i < 8; ++i) o[n][i] = 0.0f;

    float mrow[8], lrow[8];
#pragma unroll
    for (int i = 0; i < 8; ++i) { mrow[i] = -3.0e38f; lrow[i] = 0.0f; }

    const float sf = 0.08838834764831845f;   // 1/sqrt(128)
    const int   rb = qr0 + ((lane >> 4) << 3);

    for (int j = 0; j <= qb; ++j) {
        const int kb = j * 64;
        __syncthreads();

        // K tile via Tensor Data Mover: 64 rows x 256B, 8B elements, row
        // stride DMODEL*2 bytes; LDS rows padded by 16B (interval=64 DWORDs,
        // amount=4 DWORDs).  One descriptor, issued by wave 0 only.
        if (wave == 0) {
            tdm_load_2d_to_lds(
                K + (size_t)kb * DMODEL + hoff, LDS_OFF(&Ks[0][0]),
                /*tensor_d0=*/(DMODEL * 2) / 8, /*tensor_d1=*/SEQ,
                /*d0_stride=*/(DMODEL * 2) / 8,
                /*tile_d0=*/(HDIM * 2) / 8, /*tile_d1=*/64,
                /*flags=*/(3u << 16) | (1u << 20) | (5u << 22) | (3u << 25));
        }

        // V tile: read b32, scatter transposed (overlaps the TDM transfer)
        {
            int t = threadIdx.x;
#pragma unroll
            for (int c = 0; c < 32; ++c) {
                int idx = c * 128 + t;
                int row = idx >> 6;                // key 0..63
                int d2  = (idx & 63) << 1;         // dim 0..126 step 2
                unsigned int w =
                    *(const unsigned int*)(V + (size_t)(kb + row) * DMODEL + hoff + d2);
                Vt[d2][row]     = (unsigned short)(w & 0xffffu);
                Vt[d2 + 1][row] = (unsigned short)(w >> 16);
            }
        }
        if (wave == 0) __builtin_amdgcn_s_wait_tensorcnt(0);
        __syncthreads();

        // ---- S = Q * K^T (16 WMMAs: 4 key subtiles x 4 dim steps) ----
        v8f sacc[4];
#pragma unroll
        for (int s = 0; s < 4; ++s) {
#pragma unroll
            for (int i = 0; i < 8; ++i) sacc[s][i] = 0.0f;
#pragma unroll
            for (int t = 0; t < 4; ++t) {
                FragB16 bf_;
                const unsigned short* bp =
                    &Ks[s * 16 + (lane & 15)][t * 32 + ((lane >> 4) << 3)];
                bf_.u[0] = *(const v4u*)(bp);
                bf_.u[1] = *(const v4u*)(bp + 16);
                sacc[s] = wmma_bf16(qf[t], bf_, sacc[s]);
            }
        }

        // ---- scale + causal mask + online softmax (per-row, shfl in 16-lane halves)
        float alpha[8];
        const int rr = (lane >> 4) << 3;
#pragma unroll
        for (int i = 0; i < 8; ++i) {
            const int qidx = rb + i;
            float p[4], rm = -3.0e38f;
#pragma unroll
            for (int s = 0; s < 4; ++s) {
                int   kidx = kb + s * 16 + (lane & 15);
                float val  = sacc[s][i] * sf;
                val  = (kidx <= qidx) ? val : -3.0e38f;
                p[s] = val;
                rm   = fmaxf(rm, val);
            }
            rm = fmaxf(rm, __shfl_xor(rm, 1));
            rm = fmaxf(rm, __shfl_xor(rm, 2));
            rm = fmaxf(rm, __shfl_xor(rm, 4));
            rm = fmaxf(rm, __shfl_xor(rm, 8));
            float mn = fmaxf(mrow[i], rm);
            float al = __expf(mrow[i] - mn);
            float rs = 0.0f;
#pragma unroll
            for (int s = 0; s < 4; ++s) {
                float e = __expf(p[s] - mn);
                rs += e;
                Ps[wave][rr + i][s * 16 + (lane & 15)] = f2bf(e);
            }
            rs += __shfl_xor(rs, 1);
            rs += __shfl_xor(rs, 2);
            rs += __shfl_xor(rs, 4);
            rs += __shfl_xor(rs, 8);
            lrow[i]  = lrow[i] * al + rs;
            mrow[i]  = mn;
            alpha[i] = al;
        }

        // rescale running output
#pragma unroll
        for (int n = 0; n < 8; ++n)
#pragma unroll
            for (int i = 0; i < 8; ++i) o[n][i] *= alpha[i];

        // same-wave LDS store -> load fence for the P tile
        asm volatile("s_wait_dscnt 0x0" ::: "memory");

        // ---- reload P as A-fragments (16 rows x 64 keys = 2 frags of K=32) ----
        FragB16 pf[2];
        {
            const unsigned short* pp = &Ps[wave][lane & 15][(lane >> 4) << 3];
            pf[0].u[0] = *(const v4u*)(pp);
            pf[0].u[1] = *(const v4u*)(pp + 16);
            pf[1].u[0] = *(const v4u*)(pp + 32);
            pf[1].u[1] = *(const v4u*)(pp + 48);
        }

        // ---- O += P * V (16 WMMAs: 8 dim subtiles x 2 key steps) ----
#pragma unroll
        for (int n = 0; n < 8; ++n) {
#pragma unroll
            for (int kk = 0; kk < 2; ++kk) {
                FragB16 bv;
                const unsigned short* vp =
                    &Vt[n * 16 + (lane & 15)][kk * 32 + ((lane >> 4) << 3)];
                bv.u[0] = *(const v4u*)(vp);
                bv.u[1] = *(const v4u*)(vp + 16);
                o[n] = wmma_bf16(pf[kk], bv, o[n]);
            }
        }
    }

    // ---- finalize: divide by softmax denominator, store bf16 ----
#pragma unroll
    for (int n = 0; n < 8; ++n) {
        int col = hoff + n * 16 + (lane & 15);
#pragma unroll
        for (int i = 0; i < 8; ++i) {
            float v = o[n][i] / lrow[i];
            O[(size_t)(rb + i) * DMODEL + col] = f2bf(v);
        }
    }
}

// ---------------------------------------------------------------------------
// Launch: convert -> QKV GEMMs -> flash attention -> output GEMM
// ---------------------------------------------------------------------------
extern "C" void kernel_launch(void* const* d_in, const int* in_sizes, int n_in,
                              void* d_out, int out_size, void* d_ws, size_t ws_size,
                              hipStream_t stream)
{
    const float* hs = (const float*)d_in[0];
    const float* Wq = (const float*)d_in[1];
    const float* bq = (const float*)d_in[2];
    const float* Wk = (const float*)d_in[3];
    const float* bk = (const float*)d_in[4];
    const float* Wv = (const float*)d_in[5];
    const float* bv = (const float*)d_in[6];
    const float* Wo = (const float*)d_in[7];
    const float* bo = (const float*)d_in[8];
    float*      out = (float*)d_out;

    const size_t SD = (size_t)SEQ * DMODEL;      // 8 Mi elems
    const size_t DD = (size_t)DMODEL * DMODEL;   // 4 Mi elems

    unsigned short* p   = (unsigned short*)d_ws;
    unsigned short* hsb = p; p += SD;
    unsigned short* Wqb = p; p += DD;
    unsigned short* Wkb = p; p += DD;
    unsigned short* Wvb = p; p += DD;
    unsigned short* Wob = p; p += DD;
    unsigned short* qw  = p; p += SD;
    unsigned short* kw  = p; p += SD;
    unsigned short* vw  = p; p += SD;
    unsigned short* ow  = p; p += SD;

    const int cb = 256;
    f32_to_bf16_kernel<<<(int)(SD / 4 + cb - 1) / cb, cb, 0, stream>>>(hs, hsb, (int)SD);
    f32_to_bf16_kernel<<<(int)(DD / 4 + cb - 1) / cb, cb, 0, stream>>>(Wq, Wqb, (int)DD);
    f32_to_bf16_kernel<<<(int)(DD / 4 + cb - 1) / cb, cb, 0, stream>>>(Wk, Wkb, (int)DD);
    f32_to_bf16_kernel<<<(int)(DD / 4 + cb - 1) / cb, cb, 0, stream>>>(Wv, Wvb, (int)DD);
    f32_to_bf16_kernel<<<(int)(DD / 4 + cb - 1) / cb, cb, 0, stream>>>(Wo, Wob, (int)DD);

    dim3 gg(DMODEL / 128, SEQ / 128);
    gemm_bt_bias<0><<<gg, 256, 0, stream>>>(hsb, Wqb, bq, qw, SEQ, DMODEL, DMODEL);
    gemm_bt_bias<0><<<gg, 256, 0, stream>>>(hsb, Wkb, bk, kw, SEQ, DMODEL, DMODEL);
    gemm_bt_bias<0><<<gg, 256, 0, stream>>>(hsb, Wvb, bv, vw, SEQ, DMODEL, DMODEL);

    attn_fwd<<<dim3(SEQ / 64, NHEAD), 128, 0, stream>>>(qw, kw, vw, ow);

    gemm_bt_bias<1><<<gg, 256, 0, stream>>>(ow, Wob, bo, out, SEQ, DMODEL, DMODEL);
}
